// TransLayer_11244224381148
// MI455X (gfx1250) — matrix operand order repaired and checked by reading, not compiled, not run on hardware
//
#include <hip/hip_runtime.h>
#include <hip/hip_bf16.h>

// ---------------------------------------------------------------------------
// Nystromformer layer for MI455X (gfx1250, wave32, WMMA).
// All dense math via v_wmma_f32_16x16x32_bf16 (f32 accumulate).
// B matrices are always stored transposed ([N][K] row-major) so B-fragments
// are two 16B loads per lane.  Pinv chain GEMMs write their outputs
// directly in normal and/or transposed bf16 form via epilogue.
// ---------------------------------------------------------------------------

typedef __bf16 bf16_t;
typedef __attribute__((ext_vector_type(16))) __bf16   v16bf;
typedef __attribute__((ext_vector_type(8)))  float    v8f;
typedef __attribute__((ext_vector_type(4)))  unsigned u32x4;

union FragU { u32x4 q[2]; v16bf v; };

// A-fragment: 16x32 bf16 tile, row-major, row stride ldk (elements).
// lanes 0-15: M=lane, K={0..7,16..23}; lanes 16-31: M=lane-16, K={8..15,24..31}
__device__ __forceinline__ v16bf load_a_frag(const bf16_t* base, int ldk) {
  const int lane = threadIdx.x & 31;
  const int r = lane & 15, h = lane >> 4;
  FragU u;
  u.q[0] = *(const u32x4*)(base + (size_t)r * ldk + h * 8);
  u.q[1] = *(const u32x4*)(base + (size_t)r * ldk + 16 + h * 8);
  return u.v;
}

// B-fragment from B^T ([N][K] row-major, row stride ldk):
// lanes 0-15: N=lane, K={0..7,16..23}; lanes 16-31: N=lane-16, K={8..15,24..31}
__device__ __forceinline__ v16bf load_b_frag(const bf16_t* base, int ldk) {
  const int lane = threadIdx.x & 31;
  const int c = lane & 15, h = lane >> 4;
  const bf16_t* p = base + (size_t)c * ldk + h * 8;
  FragU u;
  u.q[0] = *(const u32x4*)(p);
  u.q[1] = *(const u32x4*)(p + 16);
  return u.v;
}

__device__ __forceinline__ v8f wmma_bf16(v16bf a, v16bf b, v8f c) {
  return __builtin_amdgcn_wmma_f32_16x16x32_bf16(false, a, false, b,
                                                 (short)0, c, false, false);
}

// ---------------------------------------------------------------------------
// 1. weight conversion + transpose: w_qkv[512][1536] -> wqkvT[1536][512] bf16
//    w_out[512][512] -> woutT[512][512] bf16
// ---------------------------------------------------------------------------
__global__ void cvt_weights_kernel(const float* wqkv, const float* wout,
                                   bf16_t* wqkvT, bf16_t* woutT) {
  int idx = blockIdx.x * 256 + threadIdx.x;
  if (idx < 512 * 1536) {
    int k = idx / 1536, n = idx % 1536;
    wqkvT[(size_t)n * 512 + k] = (bf16_t)wqkv[idx];
  } else {
    int j = idx - 512 * 1536;
    int k = j / 512, n = j % 512;
    woutT[(size_t)n * 512 + k] = (bf16_t)wout[j];
  }
}

// ---------------------------------------------------------------------------
// 2. LayerNorm: x[32768][512] f32 -> xn bf16
// ---------------------------------------------------------------------------
__global__ void ln_kernel(const float* __restrict__ X, const float* __restrict__ G,
                          const float* __restrict__ Bv, bf16_t* __restrict__ XN) {
  const int row = blockIdx.x;
  const int t = threadIdx.x;                 // 128 threads, 4 floats each
  const float* xr = X + (size_t)row * 512;
  float x0 = xr[t * 4 + 0], x1 = xr[t * 4 + 1], x2 = xr[t * 4 + 2], x3 = xr[t * 4 + 3];
  float s = x0 + x1 + x2 + x3;
  float ss = x0 * x0 + x1 * x1 + x2 * x2 + x3 * x3;
  for (int m = 16; m; m >>= 1) { s += __shfl_xor(s, m, 32); ss += __shfl_xor(ss, m, 32); }
  __shared__ float red[8];
  int wid = t >> 5, lane = t & 31;
  if (lane == 0) { red[wid] = s; red[4 + wid] = ss; }
  __syncthreads();
  s  = red[0] + red[1] + red[2] + red[3];
  ss = red[4] + red[5] + red[6] + red[7];
  float mu = s * (1.0f / 512.0f);
  float var = ss * (1.0f / 512.0f) - mu * mu;
  float rs = rsqrtf(var + 1e-5f);
  bf16_t* o = XN + (size_t)row * 512 + t * 4;
  o[0] = (bf16_t)((x0 - mu) * rs * G[t * 4 + 0] + Bv[t * 4 + 0]);
  o[1] = (bf16_t)((x1 - mu) * rs * G[t * 4 + 1] + Bv[t * 4 + 1]);
  o[2] = (bf16_t)((x2 - mu) * rs * G[t * 4 + 2] + Bv[t * 4 + 2]);
  o[3] = (bf16_t)((x3 - mu) * rs * G[t * 4 + 3] + Bv[t * 4 + 3]);
}

// ---------------------------------------------------------------------------
// 3. QKV GEMM: xn[32768][512] @ wqkvT -> q,k,v [B,H,N,64] bf16 (q scaled 1/8)
//    128x64 block tile, 8 waves, each wave 32x32 (2x2 WMMA quads), K step 32
// ---------------------------------------------------------------------------
__global__ void gemm_qkv_kernel(const bf16_t* __restrict__ A,
                                const bf16_t* __restrict__ BT,
                                bf16_t* __restrict__ Qo, bf16_t* __restrict__ Ko,
                                bf16_t* __restrict__ Vo) {
  const int K = 512;
  const int mbase = blockIdx.x * 128, nbase = blockIdx.y * 64;
  __shared__ __align__(16) bf16_t As[128 * 32];
  __shared__ __align__(16) bf16_t Bs[64 * 32];
  const int t = threadIdx.x;
  const int wid = t >> 5, wm = wid >> 1, wn = wid & 1;
  v8f acc[2][2] = {};
  for (int k0 = 0; k0 < K; k0 += 32) {
    {
      int row = t >> 1, seg = (t & 1) * 16;
      const bf16_t* src = A + (size_t)(mbase + row) * K + k0 + seg;
      __builtin_prefetch(src + 32, 0, 0);
      u32x4* dst = (u32x4*)&As[row * 32 + seg];
      dst[0] = ((const u32x4*)src)[0];
      dst[1] = ((const u32x4*)src)[1];
    }
    if (t < 128) {
      int n = t >> 1, seg = (t & 1) * 16;
      const bf16_t* src = BT + (size_t)(nbase + n) * K + k0 + seg;
      u32x4* dst = (u32x4*)&Bs[n * 32 + seg];
      dst[0] = ((const u32x4*)src)[0];
      dst[1] = ((const u32x4*)src)[1];
    }
    __syncthreads();
    v16bf a0 = load_a_frag(&As[(wm * 32) * 32], 32);
    v16bf a1 = load_a_frag(&As[(wm * 32 + 16) * 32], 32);
    v16bf b0 = load_b_frag(&Bs[(wn * 32) * 32], 32);
    v16bf b1 = load_b_frag(&Bs[(wn * 32 + 16) * 32], 32);
    acc[0][0] = wmma_bf16(a0, b0, acc[0][0]);
    acc[0][1] = wmma_bf16(a0, b1, acc[0][1]);
    acc[1][0] = wmma_bf16(a1, b0, acc[1][0]);
    acc[1][1] = wmma_bf16(a1, b1, acc[1][1]);
    __syncthreads();
  }
  const int hh = (t >> 4) & 1, cc = t & 15;
#pragma unroll
  for (int ti = 0; ti < 2; ++ti)
#pragma unroll
    for (int tj = 0; tj < 2; ++tj)
#pragma unroll
      for (int i = 0; i < 8; ++i) {
        int row = mbase + wm * 32 + ti * 16 + i + hh * 8;
        int col = nbase + wn * 32 + tj * 16 + cc;
        int b = row >> 13, n = row & 8191;
        int which = col >> 9, rem = col & 511;
        int h = rem >> 6, d = rem & 63;
        size_t dst = ((size_t)(b * 8 + h) * 8192 + n) * 64 + d;
        float v = acc[ti][tj][i];
        if (which == 0)      Qo[dst] = (bf16_t)(v * 0.125f);
        else if (which == 1) Ko[dst] = (bf16_t)v;
        else                 Vo[dst] = (bf16_t)v;
      }
}

// ---------------------------------------------------------------------------
// 4. Landmarks: mean over groups of 32 tokens -> ql,kl [B,H,256,64] bf16
// ---------------------------------------------------------------------------
__global__ void landmarks_kernel(const bf16_t* __restrict__ Q,
                                 const bf16_t* __restrict__ Kv,
                                 bf16_t* __restrict__ QL, bf16_t* __restrict__ KL) {
  int bh = blockIdx.x >> 8, m = blockIdx.x & 255, d = threadIdx.x;
  float sq = 0.f, sk = 0.f;
  size_t base = ((size_t)bh * 8192 + m * 32) * 64 + d;
  for (int tkn = 0; tkn < 32; ++tkn) {
    sq += (float)Q[base + (size_t)tkn * 64];
    sk += (float)Kv[base + (size_t)tkn * 64];
  }
  QL[(size_t)bh * 16384 + m * 64 + d] = (bf16_t)(sq * 0.03125f);
  KL[(size_t)bh * 16384 + m * 64 + d] = (bf16_t)(sk * 0.03125f);
}

// ---------------------------------------------------------------------------
// 5. a = softmax(Q @ KL^T) over 256 cols.  One wave = 16 rows, full 16x256
//    score strip in registers (16 WMMA C tiles), shuffle-reduced softmax.
// ---------------------------------------------------------------------------
__global__ void attn_softmax_kernel(const bf16_t* __restrict__ Q,
                                    const bf16_t* __restrict__ KL,
                                    bf16_t* __restrict__ OUT, int rows) {
  const int bh = blockIdx.y;
  const int wid = threadIdx.x >> 5;
  const int mbase = (blockIdx.x * 4 + wid) * 16;
  __shared__ __align__(16) bf16_t kls[256 * 64];          // 32 KB
  for (int seg = threadIdx.x; seg < 1024; seg += 128) {   // 1024 segs x 8 elems = 8192 of 16384 elems
    ((u32x4*)kls)[seg] = ((const u32x4*)(KL + (size_t)bh * 16384))[seg];
  }
  __syncthreads();

  const bf16_t* qb = Q + (size_t)bh * rows * 64 + (size_t)mbase * 64;
  v16bf aq0 = load_a_frag(qb, 64);        // K 0..31
  v16bf aq1 = load_a_frag(qb + 32, 64);   // K 32..63
  v8f acc[16];
#pragma unroll
  for (int tt = 0; tt < 16; ++tt) {
    v16bf b0 = load_b_frag(&kls[(tt * 16) * 64], 64);
    v16bf b1 = load_b_frag(&kls[(tt * 16) * 64 + 32], 64);
    v8f c = {};
    c = wmma_bf16(aq0, b0, c);
    acc[tt] = wmma_bf16(aq1, b1, c);
  }
  float rmax[8], rsum[8];
#pragma unroll
  for (int i = 0; i < 8; ++i) rmax[i] = -3.0e38f;
#pragma unroll
  for (int tt = 0; tt < 16; ++tt)
#pragma unroll
    for (int i = 0; i < 8; ++i) rmax[i] = fmaxf(rmax[i], acc[tt][i]);
#pragma unroll
  for (int i = 0; i < 8; ++i) {
    for (int m = 1; m < 16; m <<= 1) rmax[i] = fmaxf(rmax[i], __shfl_xor(rmax[i], m, 32));
    rsum[i] = 0.f;
  }
#pragma unroll
  for (int tt = 0; tt < 16; ++tt)
#pragma unroll
    for (int i = 0; i < 8; ++i) {
      float e = __expf(acc[tt][i] - rmax[i]);
      acc[tt][i] = e;
      rsum[i] += e;
    }
#pragma unroll
  for (int i = 0; i < 8; ++i) {
    for (int m = 1; m < 16; m <<= 1) rsum[i] += __shfl_xor(rsum[i], m, 32);
    rsum[i] = 1.0f / rsum[i];
  }
  const int hh = (threadIdx.x >> 4) & 1, cc = threadIdx.x & 15;
  bf16_t* ob = OUT + (size_t)bh * rows * 256;
#pragma unroll
  for (int tt = 0; tt < 16; ++tt)
#pragma unroll
    for (int i = 0; i < 8; ++i) {
      int row = mbase + i + hh * 8, col = tt * 16 + cc;
      ob[(size_t)row * 256 + col] = (bf16_t)(acc[tt][i] * rsum[i]);
    }
}

// ---------------------------------------------------------------------------
// 6. pinv init scalars: global max of |a2| row-sums and col-sums (atomic max
//    on positive-float bits)
// ---------------------------------------------------------------------------
__global__ void stats_zero_kernel(unsigned* s) { s[threadIdx.x] = 0u; }

__global__ void a2_stats_kernel(const bf16_t* __restrict__ A2, unsigned* mb, int tr) {
  int bh = blockIdx.x >> 8, i = blockIdx.x & 255;
  const bf16_t* base = A2 + (size_t)bh * 65536;
  float s = 0.f;
  for (int j = threadIdx.x * 4; j < 256; j += 256) {
#pragma unroll
    for (int u = 0; u < 4; ++u) {
      int jj = j + u;
      float v = tr ? (float)base[(size_t)jj * 256 + i] : (float)base[(size_t)i * 256 + jj];
      s += fabsf(v);
    }
  }
  for (int m = 16; m; m >>= 1) s += __shfl_xor(s, m, 32);
  __shared__ float red[2];
  if ((threadIdx.x & 31) == 0) red[threadIdx.x >> 5] = s;
  __syncthreads();
  if (threadIdx.x == 0) {
    float tot = red[0] + red[1];
    atomicMax(&mb[tr], __float_as_uint(tot));
  }
}

// 7. z0 = a2^T / (col*row); writes z normal and transposed bf16 forms
__global__ void z_init_kernel(const bf16_t* __restrict__ A2, const unsigned* mb,
                              bf16_t* __restrict__ zn, bf16_t* __restrict__ zt) {
  int bh = blockIdx.x >> 8, i = blockIdx.x & 255, j = threadIdx.x;
  float s = 1.0f / (__uint_as_float(mb[0]) * __uint_as_float(mb[1]));
  const bf16_t* base = A2 + (size_t)bh * 65536;
  size_t o = (size_t)bh * 65536 + (size_t)i * 256 + j;
  zn[o] = (bf16_t)((float)base[(size_t)j * 256 + i] * s);
  zt[o] = (bf16_t)((float)base[(size_t)i * 256 + j] * s);
}

// ---------------------------------------------------------------------------
// 8. Batched GEMM (32 batches), C = A[M][K] @ B ( B given as BT[N][K] ).
//    Dual epilogue:  out1[n-form] = e1*acc + s1*I ;  out2[t-form] = e2*acc + s2*I
//    outmode 1: scatter out1 into attn layout [B,N,512] (batch = b*8+h)
// ---------------------------------------------------------------------------
__global__ void gemm_bat_kernel(const bf16_t* __restrict__ A, unsigned long long sA,
                                const bf16_t* __restrict__ BT, unsigned long long sBT,
                                int K,
                                int ld1, bf16_t* __restrict__ out1, unsigned long long sO1,
                                float e1, float s1,
                                int ld2, bf16_t* __restrict__ out2, unsigned long long sO2,
                                float e2, float s2, int outmode) {
  const int batch = blockIdx.z;
  const int mbase = blockIdx.x * 128, nbase = blockIdx.y * 64;
  A  += (size_t)batch * sA;
  BT += (size_t)batch * sBT;
  __shared__ __align__(16) bf16_t As[128 * 32];
  __shared__ __align__(16) bf16_t Bs[64 * 32];
  const int t = threadIdx.x;
  const int wid = t >> 5, wm = wid >> 1, wn = wid & 1;
  v8f acc[2][2] = {};
  for (int k0 = 0; k0 < K; k0 += 32) {
    {
      int row = t >> 1, seg = (t & 1) * 16;
      const bf16_t* src = A + (size_t)(mbase + row) * K + k0 + seg;
      __builtin_prefetch(src + 32, 0, 0);
      u32x4* dst = (u32x4*)&As[row * 32 + seg];
      dst[0] = ((const u32x4*)src)[0];
      dst[1] = ((const u32x4*)src)[1];
    }
    if (t < 128) {
      int n = t >> 1, seg = (t & 1) * 16;
      const bf16_t* src = BT + (size_t)(nbase + n) * K + k0 + seg;
      u32x4* dst = (u32x4*)&Bs[n * 32 + seg];
      dst[0] = ((const u32x4*)src)[0];
      dst[1] = ((const u32x4*)src)[1];
    }
    __syncthreads();
    v16bf a0 = load_a_frag(&As[(wm * 32) * 32], 32);
    v16bf a1 = load_a_frag(&As[(wm * 32 + 16) * 32], 32);
    v16bf b0 = load_b_frag(&Bs[(wn * 32) * 32], 32);
    v16bf b1 = load_b_frag(&Bs[(wn * 32 + 16) * 32], 32);
    acc[0][0] = wmma_bf16(a0, b0, acc[0][0]);
    acc[0][1] = wmma_bf16(a0, b1, acc[0][1]);
    acc[1][0] = wmma_bf16(a1, b0, acc[1][0]);
    acc[1][1] = wmma_bf16(a1, b1, acc[1][1]);
    __syncthreads();
  }
  const int hh = (t >> 4) & 1, cc = t & 15;
#pragma unroll
  for (int ti = 0; ti < 2; ++ti)
#pragma unroll
    for (int tj = 0; tj < 2; ++tj)
#pragma unroll
      for (int i = 0; i < 8; ++i) {
        int row = mbase + wm * 32 + ti * 16 + i + hh * 8;
        int col = nbase + wn * 32 + tj * 16 + cc;
        float v = acc[ti][tj][i];
        float idn = (row == col) ? 1.0f : 0.0f;
        if (out1) {
          float o1 = e1 * v + s1 * idn;
          if (outmode == 1) {
            size_t idx = ((size_t)((batch >> 3) * 8192 + row)) * 512 +
                         (size_t)(batch & 7) * 64 + col;
            out1[idx] = (bf16_t)o1;
          } else {
            out1[(size_t)batch * sO1 + (size_t)row * ld1 + col] = (bf16_t)o1;
          }
        }
        if (out2) {
          float o2 = e2 * v + s2 * idn;
          out2[(size_t)batch * sO2 + (size_t)col * ld2 + row] = (bf16_t)o2;
        }
      }
}

// ---------------------------------------------------------------------------
// 9. flash a3v: softmax(QL @ K^T over N=8192) @ V, per (bh, 16-landmark tile).
//    Online softmax; P converted C-layout -> A-layout through LDS; V chunk
//    transposed into LDS for B-fragments.  Output written transposed [64][256].
// ---------------------------------------------------------------------------
__global__ void flash_a3v_kernel(const bf16_t* __restrict__ QL,
                                 const bf16_t* __restrict__ Kv,
                                 const bf16_t* __restrict__ V,
                                 bf16_t* __restrict__ A3VT) {
  const int mtile = blockIdx.x, bh = blockIdx.y;
  const int lane = threadIdx.x & 31;
  const int hh = lane >> 4, cc = lane & 15;
  __shared__ __align__(16) bf16_t pls[16 * 32];
  __shared__ __align__(16) bf16_t vts[64 * 32];

  const bf16_t* qb = QL + (size_t)bh * 16384 + (size_t)mtile * 16 * 64;
  v16bf aq0 = load_a_frag(qb, 64);
  v16bf aq1 = load_a_frag(qb + 32, 64);

  float runm[8], runl[8];
#pragma unroll
  for (int i = 0; i < 8; ++i) { runm[i] = -3.0e38f; runl[i] = 0.f; }
  v8f oacc[4] = {};

  const bf16_t* kb_base = Kv + (size_t)bh * 8192 * 64;
  const bf16_t* vb_base = V + (size_t)bh * 8192 * 64;

  for (int c0 = 0; c0 < 8192; c0 += 32) {
    // S = ql_tile @ k_chunk^T : 16x32 scores (2 C tiles)
    const bf16_t* kb = kb_base + (size_t)c0 * 64;
    v8f S0 = {}, S1 = {};
    S0 = wmma_bf16(aq0, load_b_frag(kb, 64), S0);
    S0 = wmma_bf16(aq1, load_b_frag(kb + 32, 64), S0);
    S1 = wmma_bf16(aq0, load_b_frag(kb + 16 * 64, 64), S1);
    S1 = wmma_bf16(aq1, load_b_frag(kb + 16 * 64 + 32, 64), S1);

    // stage V^T chunk: vts[d][t] = V[c0+t][d]
    for (int idx = lane; idx < 2048; idx += 32) {
      int tk = idx >> 6, d = idx & 63;
      vts[d * 32 + tk] = vb_base[(size_t)(c0 + tk) * 64 + d];
    }

    // online softmax update
    float corr[8];
#pragma unroll
    for (int i = 0; i < 8; ++i) {
      float tm = fmaxf(S0[i], S1[i]);
      for (int m = 1; m < 16; m <<= 1) tm = fmaxf(tm, __shfl_xor(tm, m, 32));
      float nm = fmaxf(runm[i], tm);
      corr[i] = __expf(runm[i] - nm);
      float p0 = __expf(S0[i] - nm);
      float p1 = __expf(S1[i] - nm);
      float ps = p0 + p1;
      for (int m = 1; m < 16; m <<= 1) ps += __shfl_xor(ps, m, 32);
      runl[i] = runl[i] * corr[i] + ps;
      runm[i] = nm;
      int prow = i + hh * 8;
      pls[prow * 32 + cc]      = (bf16_t)p0;
      pls[prow * 32 + 16 + cc] = (bf16_t)p1;
    }
    __syncthreads();
#pragma unroll
    for (int dt = 0; dt < 4; ++dt) {
#pragma unroll
      for (int i = 0; i < 8; ++i) oacc[dt][i] *= corr[i];
    }
    v16bf pa = load_a_frag(pls, 32);
#pragma unroll
    for (int dt = 0; dt < 4; ++dt) {
      v16bf vb = load_b_frag(&vts[(dt * 16) * 32], 32);
      oacc[dt] = wmma_bf16(pa, vb, oacc[dt]);
    }
    __syncthreads();
  }
  float invl[8];
#pragma unroll
  for (int i = 0; i < 8; ++i) invl[i] = 1.0f / runl[i];
  bf16_t* ob = A3VT + (size_t)bh * 16384;
#pragma unroll
  for (int dt = 0; dt < 4; ++dt)
#pragma unroll
    for (int i = 0; i < 8; ++i) {
      int m = mtile * 16 + i + hh * 8;
      int d = dt * 16 + cc;
      ob[(size_t)d * 256 + m] = (bf16_t)(oacc[dt][i] * invl[i]);
    }
}

// ---------------------------------------------------------------------------
// 10. depthwise conv residual: ares = attn + conv33(v)   (bf16 out)
// ---------------------------------------------------------------------------
__global__ void conv_res_kernel(const bf16_t* __restrict__ ATT,
                                const bf16_t* __restrict__ V,
                                const float* __restrict__ WC,
                                bf16_t* __restrict__ ARES) {
  size_t idx = (size_t)blockIdx.x * 256 + threadIdx.x;
  int c = idx & 511;
  int n = (idx >> 9) & 8191;
  int b = idx >> 22;
  int h = c >> 6, d = c & 63;
  float acc = (float)ATT[idx];
  size_t vbase = ((size_t)(b * 8 + h) * 8192) * 64 + d;
#pragma unroll
  for (int tap = 0; tap < 33; ++tap) {
    int nn = n + tap - 16;
    if (nn >= 0 && nn < 8192)
      acc += (float)V[vbase + (size_t)nn * 64] * WC[h * 33 + tap];
  }
  ARES[idx] = (bf16_t)acc;
}

// ---------------------------------------------------------------------------
// 11. final projection: y = x + ares @ woutT + b_out   (f32 out)
// ---------------------------------------------------------------------------
__global__ void gemm_final_kernel(const bf16_t* __restrict__ A,
                                  const bf16_t* __restrict__ BT,
                                  const float* __restrict__ Bo,
                                  const float* __restrict__ X,
                                  float* __restrict__ Y) {
  const int K = 512;
  const int mbase = blockIdx.x * 128, nbase = blockIdx.y * 64;
  __shared__ __align__(16) bf16_t As[128 * 32];
  __shared__ __align__(16) bf16_t Bs[64 * 32];
  const int t = threadIdx.x;
  const int wid = t >> 5, wm = wid >> 1, wn = wid & 1;
  v8f acc[2][2] = {};
  for (int k0 = 0; k0 < K; k0 += 32) {
    {
      int row = t >> 1, seg = (t & 1) * 16;
      const bf16_t* src = A + (size_t)(mbase + row) * K + k0 + seg;
      __builtin_prefetch(src + 32, 0, 0);
      u32x4* dst = (u32x4*)&As[row * 32 + seg];
      dst[0] = ((const u32x4*)src)[0];
      dst[1] = ((const u32x4*)src)[1];
    }
    if (t < 128) {
      int n = t >> 1, seg = (t & 1) * 16;
      const bf16_t* src = BT + (size_t)(nbase + n) * K + k0 + seg;
      u32x4* dst = (u32x4*)&Bs[n * 32 + seg];
      dst[0] = ((const u32x4*)src)[0];
      dst[1] = ((const u32x4*)src)[1];
    }
    __syncthreads();
    v16bf a0 = load_a_frag(&As[(wm * 32) * 32], 32);
    v16bf a1 = load_a_frag(&As[(wm * 32 + 16) * 32], 32);
    v16bf b0 = load_b_frag(&Bs[(wn * 32) * 32], 32);
    v16bf b1 = load_b_frag(&Bs[(wn * 32 + 16) * 32], 32);
    acc[0][0] = wmma_bf16(a0, b0, acc[0][0]);
    acc[0][1] = wmma_bf16(a0, b1, acc[0][1]);
    acc[1][0] = wmma_bf16(a1, b0, acc[1][0]);
    acc[1][1] = wmma_bf16(a1, b1, acc[1][1]);
    __syncthreads();
  }
  const int hh = (t >> 4) & 1, cc = t & 15;
#pragma unroll
  for (int ti = 0; ti < 2; ++ti)
#pragma unroll
    for (int tj = 0; tj < 2; ++tj)
#pragma unroll
      for (int i = 0; i < 8; ++i) {
        int row = mbase + wm * 32 + ti * 16 + i + hh * 8;
        int col = nbase + wn * 32 + tj * 16 + cc;
        size_t o = (size_t)row * 512 + col;
        Y[o] = acc[ti][tj][i] + Bo[col] + X[o];
      }
}

// ---------------------------------------------------------------------------
// host-side orchestration
// ---------------------------------------------------------------------------
extern "C" void kernel_launch(void* const* d_in, const int* in_sizes, int n_in,
                              void* d_out, int out_size, void* d_ws, size_t ws_size,
                              hipStream_t stream) {
  const float* x     = (const float*)d_in[0];
  const float* ln_g  = (const float*)d_in[1];
  const float* ln_b  = (const float*)d_in[2];
  const float* wqkv  = (const float*)d_in[3];
  const float* wout  = (const float*)d_in[4];
  const float* b_out = (const float*)d_in[5];
  const float* wconv = (const float*)d_in[6];
  float* y = (float*)d_out;

  char* p = (char*)d_ws;
  auto take = [&](size_t bytes) { char* r = p; p += (bytes + 255) & ~(size_t)255; return r; };

  const size_t NT = 32768;                       // B*N tokens
  bf16_t* xn    = (bf16_t*)take(NT * 512 * 2);
  bf16_t* wqkvT = (bf16_t*)take(1536 * 512 * 2);
  bf16_t* woutT = (bf16_t*)take(512 * 512 * 2);
  bf16_t* q     = (bf16_t*)take(NT * 512 * 2);   // [B,H,8192,64]
  bf16_t* k     = (bf16_t*)take(NT * 512 * 2);
  bf16_t* v     = (bf16_t*)take(NT * 512 * 2);
  bf16_t* ql    = (bf16_t*)take(32 * 256 * 64 * 2);
  bf16_t* kl    = (bf16_t*)take(32 * 256 * 64 * 2);
  bf16_t* a1    = (bf16_t*)take((size_t)32 * 8192 * 256 * 2);
  bf16_t* a2    = (bf16_t*)take(32 * 256 * 256 * 2);
  bf16_t* z1n   = (bf16_t*)take(32 * 256 * 256 * 2);
  bf16_t* z1t   = (bf16_t*)take(32 * 256 * 256 * 2);
  bf16_t* z2n   = (bf16_t*)take(32 * 256 * 256 * 2);
  bf16_t* z2t   = (bf16_t*)take(32 * 256 * 256 * 2);
  bf16_t* az    = (bf16_t*)take(32 * 256 * 256 * 2);
  bf16_t* w1t   = (bf16_t*)take(32 * 256 * 256 * 2);
  bf16_t* w2t   = (bf16_t*)take(32 * 256 * 256 * 2);
  bf16_t* w3t   = (bf16_t*)take(32 * 256 * 256 * 2);
  bf16_t* a3vt  = (bf16_t*)take(32 * 64 * 256 * 2);
  bf16_t* wmt   = (bf16_t*)take(32 * 64 * 256 * 2);
  bf16_t* attn  = (bf16_t*)take(NT * 512 * 2);
  bf16_t* ares  = (bf16_t*)take(NT * 512 * 2);
  unsigned* stats = (unsigned*)take(256);
  (void)n_in; (void)in_sizes; (void)out_size; (void)ws_size;

  // weights + layernorm
  cvt_weights_kernel<<<4096, 256, 0, stream>>>(wqkv, wout, wqkvT, woutT);
  ln_kernel<<<NT, 128, 0, stream>>>(x, ln_g, ln_b, xn);

  // qkv projection
  gemm_qkv_kernel<<<dim3(256, 24), 256, 0, stream>>>(xn, wqkvT, q, k, v);

  // landmarks
  landmarks_kernel<<<32 * 256, 64, 0, stream>>>(q, k, ql, kl);

  // a1 = softmax(q kl^T)   a2 = softmax(ql kl^T)
  attn_softmax_kernel<<<dim3(128, 32), 128, 0, stream>>>(q, kl, a1, 8192);
  attn_softmax_kernel<<<dim3(4, 32), 128, 0, stream>>>(ql, kl, a2, 256);

  // pinv init
  stats_zero_kernel<<<1, 2, 0, stream>>>(stats);
  a2_stats_kernel<<<32 * 256, 64, 0, stream>>>(a2, stats, 0);
  a2_stats_kernel<<<32 * 256, 64, 0, stream>>>(a2, stats, 1);
  z_init_kernel<<<32 * 256, 256, 0, stream>>>(a2, stats, z1n, z1t);

  // Newton-Schulz iterations (bf16 WMMA, dual n/t epilogues)
  const unsigned long long S = 65536ull;
  bf16_t *zn = z1n, *zt = z1t, *zn2 = z2n, *zt2 = z2t;
  for (int it = 0; it < 6; ++it) {
    // az = a2 @ z ;  w1^T = (7I - az)^T
    gemm_bat_kernel<<<dim3(2, 4, 32), 256, 0, stream>>>(a2, S, zt, S, 256,
        256, az, S, 1.f, 0.f, 256, w1t, S, -1.f, 7.f, 0);
    // w2^T = (15I - az@w1)^T
    gemm_bat_kernel<<<dim3(2, 4, 32), 256, 0, stream>>>(az, S, w1t, S, 256,
        0, nullptr, 0, 0.f, 0.f, 256, w2t, S, -1.f, 15.f, 0);
    // w3^T = (13I - az@w2)^T
    gemm_bat_kernel<<<dim3(2, 4, 32), 256, 0, stream>>>(az, S, w2t, S, 256,
        0, nullptr, 0, 0.f, 0.f, 256, w3t, S, -1.f, 13.f, 0);
    // z' = 0.25 * z @ w3  (both forms)
    gemm_bat_kernel<<<dim3(2, 4, 32), 256, 0, stream>>>(zn, S, w3t, S, 256,
        256, zn2, S, 0.25f, 0.f, 256, zt2, S, 0.25f, 0.f, 0);
    bf16_t* tmp;
    tmp = zn; zn = zn2; zn2 = tmp;
    tmp = zt; zt = zt2; zt2 = tmp;
  }

  // a3v^T = (softmax(ql k^T) @ v)^T   (flash, never materializes a3)
  flash_a3v_kernel<<<dim3(16, 32), 32, 0, stream>>>(ql, k, v, a3vt);

  // w^T = (z @ a3v)^T    [64][256] per bh
  gemm_bat_kernel<<<dim3(2, 1, 32), 256, 0, stream>>>(zn, S, a3vt, 16384ull, 256,
      0, nullptr, 0, 0.f, 0.f, 256, wmt, 16384ull, 1.f, 0.f, 0);

  // attn = a1 @ w  -> scattered into [B,N,512]
  gemm_bat_kernel<<<dim3(64, 1, 32), 256, 0, stream>>>(a1, 8192ull * 256ull,
      wmt, 16384ull, 256, 512, attn, 0, 1.f, 0.f, 0, nullptr, 0, 0.f, 0.f, 1);

  // + depthwise conv residual
  conv_res_kernel<<<65536, 256, 0, stream>>>(attn, v, wconv, ares);

  // y = x + ares @ w_out + b_out
  gemm_final_kernel<<<dim3(256, 8), 256, 0, stream>>>(ares, woutT, b_out, x, y);
}
